// BinaryLinear_48550310314203
// MI455X (gfx1250) — matrix operand order repaired and checked by reading, not compiled, not run on hardware
//
#include <hip/hip_runtime.h>

// MI455X / gfx1250 BinaryLinear: out[m,n] = alpha[n]*sum_k x[m,k]*sign(w[n,k]) + bias[n]
// GEMM M=16384 N=2048 K=2048 (137 GFLOP). sign(w)=+/-1 is EXACT in fp16 ->
// v_wmma_f32_16x16x32_f16 with f32 accumulation; alpha/bias in f32 epilogue.
// Roofline: HBM floor ~285MB (~12us @23.3TB/s); f16 WMMA compute is the bound.
// Pre-convert x to f16 once (x_h 67MB + w_h 8.4MB fit in 192MB L2), and
// double-buffered LDS tiling (128x128x32) fed by GLOBAL_LOAD_ASYNC_TO_LDS
// (ASYNCcnt path) so each block touches each A/B byte exactly once per K-slice
// and the hot loop is ds_load_b128 + v_wmma only (no cvt chains).

typedef _Float16 half_t;
typedef __attribute__((ext_vector_type(16))) _Float16 v16h;
typedef __attribute__((ext_vector_type(8)))  _Float16 v8h;
typedef __attribute__((ext_vector_type(8)))  float    v8f;
typedef __attribute__((ext_vector_type(4)))  float    v4f;
typedef int bl_v4i __attribute__((vector_size(16)));   // builtin's pointee type

// ---- CDNA5 async global->LDS copy (guarded: falls back to load+ds_store) ----
#if defined(__gfx1250__) && __has_builtin(__builtin_amdgcn_global_load_async_to_lds_b128)
#define BL_HAS_ASYNC 1
#else
#define BL_HAS_ASYNC 0
#endif

__device__ __forceinline__ void bl_copy16(const half_t* g, half_t* l) {
#if BL_HAS_ASYNC
    __builtin_amdgcn_global_load_async_to_lds_b128(
        (__attribute__((address_space(1))) bl_v4i*)(uintptr_t)g,
        (__attribute__((address_space(3))) bl_v4i*)(uintptr_t)l,
        0, 0);
#else
    *(v8h*)l = *(const v8h*)g;
#endif
}

__device__ __forceinline__ void bl_wait_async() {
#if BL_HAS_ASYNC
#if __has_builtin(__builtin_amdgcn_s_wait_asynccnt)
    __builtin_amdgcn_s_wait_asynccnt(0);
#else
    asm volatile("s_wait_asynccnt 0" ::: "memory");
#endif
#endif
}

// ---------------------------------------------------------------------------
// Kernel 1: alpha[o] = mean|w[o,:]|, w_h[o,k] = sign(w) as fp16 +/-1.
// ---------------------------------------------------------------------------
__global__ __launch_bounds__(256)
void binlin_prep(const float* __restrict__ w, half_t* __restrict__ wh,
                 float* __restrict__ alpha, int K) {
    __shared__ float red[256];
    const int o = blockIdx.x;
    const float* row = w + (size_t)o * K;
    half_t* hrow = wh + (size_t)o * K;

    float s = 0.0f;
    for (int i = threadIdx.x; i < K; i += 256) {
        float v = row[i];
        s += fabsf(v);
        hrow[i] = (v >= 0.0f) ? (half_t)1.0f : (half_t)(-1.0f);
    }
    red[threadIdx.x] = s;
    __syncthreads();
    #pragma unroll
    for (int off = 128; off > 0; off >>= 1) {
        if (threadIdx.x < off) red[threadIdx.x] += red[threadIdx.x + off];
        __syncthreads();
    }
    if (threadIdx.x == 0) alpha[o] = red[0] * (1.0f / (float)K);
}

// ---------------------------------------------------------------------------
// Kernel 2: x (f32) -> x_h (f16), 8 elems/thread, b128 loads.
// ---------------------------------------------------------------------------
__global__ __launch_bounds__(256)
void binlin_cvt_x(const float* __restrict__ x, half_t* __restrict__ xh, int total8) {
    int i = blockIdx.x * 256 + threadIdx.x;
    if (i >= total8) return;
    size_t base = (size_t)i * 8;
    v4f f0 = *(const v4f*)(x + base);
    v4f f1 = *(const v4f*)(x + base + 4);
    v8h h;
    #pragma unroll
    for (int e = 0; e < 4; ++e) { h[e] = (half_t)f0[e]; h[4 + e] = (half_t)f1[e]; }
    *(v8h*)(xh + base) = h;
}

// ---------------------------------------------------------------------------
// Kernel 3: WMMA GEMM, block=256 (8 wave32), tile 128x128, BK=32,
// double-buffered LDS. Wave grid 4(M) x 2(N); wave = 2x4 16x16 C tiles.
//
// Per-lane fragment layouts per CDNA5 ISA 7.12.2 (wave32):
//   A 16x32 f16:  lane&15 = row M; halves 0..7 = K[kb..kb+7],
//                 halves 8..15 = K[kb+16..kb+23], kb=(lane>>4)*8
//   B 32x16 f16:  lane&15 = col N; same K order as A (kb=(lane>>4)*8)
//   C/D 16x16 f32: VGPR r -> M = r + (lane>>4)*8, N = lane&15
// LDS row stride 40 halves (80B = 20 banks) -> conflict-free ds_load_b128.
// ---------------------------------------------------------------------------
template <bool XH>
__global__ __launch_bounds__(256)
void binlin_gemm(const float* __restrict__ x, const half_t* __restrict__ xh,
                 const half_t* __restrict__ wh,
                 const float* __restrict__ alpha, const float* __restrict__ bias,
                 float* __restrict__ out, int M, int N, int K) {
    constexpr int BK  = 32;
    constexpr int LDA = BK + 8;                    // 40 halves
    __shared__ half_t Ab[2][128 * LDA];            // 20 KB
    __shared__ half_t Bb[2][128 * LDA];            // 20 KB

    const int tid   = threadIdx.x;
    const int lane  = tid & 31;
    const int wave  = tid >> 5;
    const int waveM = wave & 3;
    const int waveN = wave >> 2;
    const int l15   = lane & 15;
    const int hi    = lane >> 4;

    const int mBlk = blockIdx.y * 128;
    const int nBlk = blockIdx.x * 128;

    // Copy mapping: tile = 128 rows x 32 halves = 512 x 16B chunks; 2/thread.
    const int c0 = tid, c1 = tid + 256;
    const int r0 = c0 >> 2, k0c = (c0 & 3) * 8;
    const int r1 = c1 >> 2, k1c = (c1 & 3) * 8;

    v8f acc[2][4];
    #pragma unroll
    for (int i = 0; i < 2; ++i)
        #pragma unroll
        for (int j = 0; j < 4; ++j) { v8f z = {}; acc[i][j] = z; }

    auto load_stage = [&](int k0, int buf) {
        // A tile (f16 direct, or f32 source converted in-flight)
        if (XH) {
            bl_copy16(xh + (size_t)(mBlk + r0) * K + k0 + k0c, &Ab[buf][r0 * LDA + k0c]);
            bl_copy16(xh + (size_t)(mBlk + r1) * K + k0 + k1c, &Ab[buf][r1 * LDA + k1c]);
        } else {
            const float* g0 = x + (size_t)(mBlk + r0) * K + k0 + k0c;
            const float* g1 = x + (size_t)(mBlk + r1) * K + k0 + k1c;
            v4f a0 = *(const v4f*)g0, a1 = *(const v4f*)(g0 + 4);
            v4f b0 = *(const v4f*)g1, b1 = *(const v4f*)(g1 + 4);
            v8h h0, h1;
            #pragma unroll
            for (int e = 0; e < 4; ++e) {
                h0[e] = (half_t)a0[e]; h0[4 + e] = (half_t)a1[e];
                h1[e] = (half_t)b0[e]; h1[4 + e] = (half_t)b1[e];
            }
            *(v8h*)&Ab[buf][r0 * LDA + k0c] = h0;
            *(v8h*)&Ab[buf][r1 * LDA + k1c] = h1;
        }
        // B tile (fp16 +/-1)
        bl_copy16(wh + (size_t)(nBlk + r0) * K + k0 + k0c, &Bb[buf][r0 * LDA + k0c]);
        bl_copy16(wh + (size_t)(nBlk + r1) * K + k0 + k1c, &Bb[buf][r1 * LDA + k1c]);
    };

    const int ar0   = waveM * 32 + l15;
    const int bn0   = waveN * 64 + l15;
    const int kaOff = hi * 8;
    const int kbOff = hi * 8;

    load_stage(0, 0);
    bl_wait_async();
    __syncthreads();

    int buf = 0;
    #pragma unroll 1
    for (int k0 = 0; k0 < K; k0 += BK) {
        if (k0 + BK < K) load_stage(k0 + BK, buf ^ 1);

        v16h a[2], b[4];
        #pragma unroll
        for (int i = 0; i < 2; ++i) {
            const half_t* p = &Ab[buf][(ar0 + i * 16) * LDA + kaOff];
            v8h lo = *(const v8h*)p;
            v8h hh = *(const v8h*)(p + 16);
            #pragma unroll
            for (int e = 0; e < 8; ++e) { a[i][e] = lo[e]; a[i][8 + e] = hh[e]; }
        }
        #pragma unroll
        for (int j = 0; j < 4; ++j) {
            const half_t* p = &Bb[buf][(bn0 + j * 16) * LDA + kbOff];
            v8h lo = *(const v8h*)p;
            v8h hh = *(const v8h*)(p + 16);
            #pragma unroll
            for (int e = 0; e < 8; ++e) { b[j][e] = lo[e]; b[j][8 + e] = hh[e]; }
        }
        #pragma unroll
        for (int i = 0; i < 2; ++i)
            #pragma unroll
            for (int j = 0; j < 4; ++j)
                acc[i][j] = __builtin_amdgcn_wmma_f32_16x16x32_f16(
                    false, a[i], false, b[j], (short)0, acc[i][j], false, false);

        bl_wait_async();
        __syncthreads();
        buf ^= 1;
    }

    // Epilogue: out = alpha[n]*acc + bias[n] (f32), coalesced over lanes 0..15.
    const int m0 = mBlk + waveM * 32;
    const int n0 = nBlk + waveN * 64;
    #pragma unroll
    for (int j = 0; j < 4; ++j) {
        const int n   = n0 + j * 16 + l15;
        const float al = alpha[n];
        const float bi = bias[n];
        #pragma unroll
        for (int i = 0; i < 2; ++i) {
            const int mbase = m0 + i * 16 + hi * 8;
            #pragma unroll
            for (int r = 0; r < 8; ++r)
                out[(size_t)(mbase + r) * N + n] = al * acc[i][j][r] + bi;
        }
    }
}

// ---------------------------------------------------------------------------
extern "C" void kernel_launch(void* const* d_in, const int* in_sizes, int n_in,
                              void* d_out, int out_size, void* d_ws, size_t ws_size,
                              hipStream_t stream) {
    const float* x    = (const float*)d_in[0];   // [M, K]
    const float* w    = (const float*)d_in[1];   // [N, K]
    const float* bias = (const float*)d_in[2];   // [N]
    float* out        = (float*)d_out;           // [M, N]

    const int N = in_sizes[2];
    const int K = in_sizes[1] / N;
    const int M = in_sizes[0] / K;               // 16384

    // Workspace: alpha[N] f32 | w_h[N*K] f16 | (optional) x_h[M*K] f16.
    float*  alpha = (float*)d_ws;
    size_t  offWh = ((size_t)N * sizeof(float) + 255) & ~(size_t)255;
    half_t* wh    = (half_t*)((char*)d_ws + offWh);
    size_t  offXh = (offWh + (size_t)N * K * sizeof(half_t) + 255) & ~(size_t)255;
    half_t* xhp   = (half_t*)((char*)d_ws + offXh);
    const bool use_xh = ws_size >= offXh + (size_t)M * K * sizeof(half_t);

    binlin_prep<<<N, 256, 0, stream>>>(w, wh, alpha, K);

    dim3 grid(N / 128, M / 128);
    if (use_xh) {
        const int total8 = (int)(((size_t)M * K) / 8);
        binlin_cvt_x<<<(total8 + 255) / 256, 256, 0, stream>>>(x, xhp, total8);
        binlin_gemm<true><<<grid, 256, 0, stream>>>(x, xhp, wh, alpha, bias, out, M, N, K);
    } else {
        binlin_gemm<false><<<grid, 256, 0, stream>>>(x, nullptr, wh, alpha, bias, out, M, N, K);
    }
}
